// Main_11175504904192
// MI455X (gfx1250) — matrix operand order repaired and checked by reading, not compiled, not run on hardware
//
#include <hip/hip_runtime.h>

// CDNA5 / gfx1250 fused softmin-cdist kernel.
// pred = softmax(-cdist(X, Addr)/beta) @ M, never materializing the [N,A] matrix.
// Both GEMMs (X@Addr^T and w@M) run on v_wmma_f32_16x16x32_bf16.
// Round 3: drop the full s_wait_dscnt 0 drain (per-wave LDS ops are in-order;
// compiler inserts exact partial dscnt waits for register deps), keeping only
// a compiler-level fence so the transpose stage overlaps with global loads.

typedef __attribute__((ext_vector_type(16))) __bf16 v16bf;
typedef __attribute__((ext_vector_type(2)))  __bf16 bf16x2;
typedef __attribute__((ext_vector_type(8)))  float  v8f;

#define N_     4096
#define A_     32768
#define F_     160      // 5 K-blocks of 32
#define C_     10
#define SPLITA 16       // parallel splits over the A dimension
#define LDST   34       // LDS row stride in floats (even, conflict-free)

// ws layout (bytes):
//   addrB : [A_*F_] bf16                      @ 0          (10.5 MB)
//   mext  : [A_/32][32 lanes][16] bf16        @ 10485760   (1 MB)
//   y2    : [A_] f32                          @ 11534336   (128 KB)
//   partial: [SPLITA][N_][16] f32             @ 11665408   (4 MB)

// ---------------- prologue: Address f32 -> bf16 row-major -------------------
__global__ __launch_bounds__(256) void cvt_addr_kernel(
    const float* __restrict__ Addr, __bf16* __restrict__ out) {
  const size_t tid = (size_t)blockIdx.x * 256 + threadIdx.x;  // one per pair
  const float2 q = *(const float2*)(Addr + 2 * tid);
  bf16x2 p; p.x = (__bf16)q.x; p.y = (__bf16)q.y;
  *(bf16x2*)(out + 2 * tid) = p;                              // 4B store
}

// ---------------- prologue: Mext in exact B-fragment layout -----------------
// For 32-row block b, lane l (n=l&15, h=l>>4), slot j2: value = Mext[b*32+h*16+j2][n]
// where Mext cols: 0..9 = M, 10 = 1.0 (Z column), 11..15 = 0.
__global__ __launch_bounds__(256) void mext_kernel(
    const float* __restrict__ M, __bf16* __restrict__ mext) {
  const int tid = blockIdx.x * 256 + threadIdx.x;             // A_*16 total
  const int j2 = tid & 15;
  const int l  = (tid >> 4) & 31;
  const int b  = tid >> 9;
  const int n = l & 15, h = l >> 4;
  const int a = b * 32 + h * 16 + j2;
  const float v = (n < C_) ? M[(size_t)a * C_ + n] : ((n == C_) ? 1.f : 0.f);
  mext[tid] = (__bf16)v;
}

// ---------------- prologue: y2[a] = ||Address[a]||^2 ------------------------
__global__ __launch_bounds__(256) void y2_kernel(const float* __restrict__ Addr,
                                                 float* __restrict__ y2) {
  const int wave = threadIdx.x >> 5, lane = threadIdx.x & 31;
  const int row = blockIdx.x * 8 + wave;
  const float* p = Addr + (size_t)row * F_;
  float s = 0.f;
#pragma unroll
  for (int j = 0; j < 5; ++j) { float v = p[lane + 32 * j]; s += v * v; }
#pragma unroll
  for (int off = 16; off >= 1; off >>= 1) s += __shfl_xor(s, off, 32);
  if (lane == 0) y2[row] = s;
}

// ---------------- main fused kernel ----------------------------------------
// Each wave owns TWO 16-row tiles of X (r0 .. r0+31) and walks a 2048-wide
// slice of A in steps of 32: per step, 2x(5 WMMA score) per B tile-pair
// (= 20 WMMA), epilogue exp(-dist/beta) through LDS, then 2 accumulation
// WMMAs acc[u] += w_u[16x32] @ Mext[32x16].
__global__ __launch_bounds__(128) void dist_softmin_kernel(
    const float* __restrict__ X, const __bf16* __restrict__ AddrB,
    const __bf16* __restrict__ Mext, const float* __restrict__ y2,
    float* __restrict__ partial) {
  __shared__ float smem[4][2][16 * LDST];
  const int lane = threadIdx.x & 31;
  const int wave = threadIdx.x >> 5;
  const int h    = lane >> 4;      // lane half
  const int nl   = lane & 15;
  const int r0   = blockIdx.x * 128 + wave * 32;

  // --- load both X tiles as bf16 A-fragments; fold in sum-of-squares ------
  // A-fragment k order used here (16x32 bf16): lanes 0-15 row=lane, K in {0..15};
  // lanes 16-31 row=lane-16, K in {16..31}.
  v16bf afrag[2][5];
  float x2v[2][8];
#pragma unroll
  for (int u = 0; u < 2; ++u) {
    float x2part = 0.f;
    const float* xrow = X + (size_t)(r0 + u * 16 + nl) * F_;
#pragma unroll
    for (int blk = 0; blk < 5; ++blk) {
#pragma unroll
      for (int j = 0; j < 8; ++j) {
        const int k0 = 2 * j + 16 * h + blk * 32;
        const float f0 = xrow[k0], f1 = xrow[k0 + 1];
        x2part += f0 * f0 + f1 * f1;
        afrag[u][blk][2 * j]     = (__bf16)f0;
        afrag[u][blk][2 * j + 1] = (__bf16)f1;
      }
    }
    const float x2row = x2part + __shfl_xor(x2part, 16, 32);
#pragma unroll
    for (int i = 0; i < 8; ++i) x2v[u][i] = __shfl(x2row, i + 8 * h, 32);
  }

  v8f acc[2] = {{}, {}};           // cols 0..9 = sum w*M, col 10 = Z
  const float negInvBetaLog2e = -1.4426950408889634f / 1.1f;

  const int aBegin = blockIdx.y * (A_ / SPLITA);
  const int aEnd   = aBegin + (A_ / SPLITA);
  for (int a0 = aBegin; a0 < aEnd; a0 += 32) {
    // Mext fragment: single aligned 32B load, shared by both row tiles
    const v16bf mfrag = *(const v16bf*)(Mext + ((size_t)a0 + lane) * 16);

#pragma unroll
    for (int t = 0; t < 2; ++t) {
      const int acol = a0 + t * 16 + nl;
      // B-fragment, same k order as A: lane col = nl, lanes 0-15 K=0..15, lanes 16-31 K=16..31;
      // 16 contiguous bf16 per fragment slice (32B aligned).
      const __bf16* brow = AddrB + (size_t)acol * F_ + h * 16;
      v16bf bfrag[5];
#pragma unroll
      for (int blk = 0; blk < 5; ++blk)
        bfrag[blk] = *(const v16bf*)(brow + blk * 32);
      const float y2a = y2[acol];

#pragma unroll
      for (int u = 0; u < 2; ++u) {
        v8f sv = {};
#pragma unroll
        for (int blk = 0; blk < 5; ++blk)
          sv = __builtin_amdgcn_wmma_f32_16x16x32_bf16(
              false, afrag[u][blk], false, bfrag[blk], (short)0, sv, false,
              false);
#pragma unroll
        for (int i = 0; i < 8; ++i) {
          float d2 = fmaf(-2.f, sv[i], x2v[u][i]) + y2a;
          d2 = fmaxf(d2, 0.f);
          const float w = exp2f(__fsqrt_rn(d2) * negInvBetaLog2e);
          smem[wave][u][(i + 8 * h) * LDST + t * 16 + nl] = w;
        }
      }
    }
    // Per-wave LDS ops are processed in order by the LDS pipe (ISA 7.3):
    // the ds_loads below will observe the ds_stores above without a full
    // dscnt drain. Compiler-level fence only, so scheduling stays free.
    asm volatile("" ::: "memory");

#pragma unroll
    for (int u = 0; u < 2; ++u) {
      // re-read w as bf16 A-fragment (16 rows x 32 K)
      v16bf wfrag;
      const float* lds = smem[wave][u];
#pragma unroll
      for (int j = 0; j < 8; ++j) {
        const int k0 = 2 * j + 16 * h;
        const float2 p = *(const float2*)(lds + nl * LDST + k0);
        wfrag[2 * j]     = (__bf16)p.x;
        wfrag[2 * j + 1] = (__bf16)p.y;
      }
      acc[u] = __builtin_amdgcn_wmma_f32_16x16x32_bf16(
          false, wfrag, false, mfrag, (short)0, acc[u], false, false);
    }
  }

  // ---- spill per-split partial tiles: [SPLITA][N][16] ---------------------
#pragma unroll
  for (int u = 0; u < 2; ++u) {
    float* pp = partial + ((size_t)blockIdx.y * N_ + r0 + u * 16) * 16;
#pragma unroll
    for (int i = 0; i < 8; ++i) pp[(i + 8 * h) * 16 + nl] = acc[u][i];
  }
}

// ---------------- finalize: combine splits, divide by Z ---------------------
__global__ __launch_bounds__(256) void finalize_kernel(
    const float* __restrict__ partial, float* __restrict__ out) {
  const int idx = blockIdx.x * 256 + threadIdx.x;
  if (idx >= N_ * C_) return;
  const int n = idx / C_, c = idx % C_;
  float s = 0.f, z = 0.f;
#pragma unroll
  for (int sp = 0; sp < SPLITA; ++sp) {
    const float* p = partial + ((size_t)sp * N_ + n) * 16;
    s += p[c];
    z += p[C_];
  }
  out[idx] = s / z;
}

extern "C" void kernel_launch(void* const* d_in, const int* in_sizes, int n_in,
                              void* d_out, int out_size, void* d_ws, size_t ws_size,
                              hipStream_t stream) {
  const float* X    = (const float*)d_in[0];   // [N, F]
  const float* Addr = (const float*)d_in[1];   // [A, F]
  const float* M    = (const float*)d_in[2];   // [A, C]
  float* out = (float*)d_out;                  // [N, C]

  char* ws = (char*)d_ws;
  __bf16* addrB   = (__bf16*)(ws);                         // 10.5 MB
  __bf16* mext    = (__bf16*)(ws + 10485760);              // 1 MB
  float*  y2      = (float*)(ws + 11534336);               // 128 KB
  float*  partial = (float*)(ws + 11665408);               // 4 MB

  cvt_addr_kernel<<<(A_ * F_ / 2) / 256, 256, 0, stream>>>(Addr, addrB);
  mext_kernel<<<(A_ * 16) / 256, 256, 0, stream>>>(M, mext);
  y2_kernel<<<A_ / 8, 256, 0, stream>>>(Addr, y2);
  dist_softmin_kernel<<<dim3(N_ / 128, SPLITA), 128, 0, stream>>>(
      X, addrB, mext, y2, partial);
  finalize_kernel<<<(N_ * C_ + 255) / 256, 256, 0, stream>>>(partial, out);
}
